// Decoder_56934086475838
// MI455X (gfx1250) — matrix operand order repaired and checked by reading, not compiled, not run on hardware
//
#include <hip/hip_runtime.h>
#include <hip/hip_bf16.h>

#define DEVINL __device__ __forceinline__

typedef __attribute__((ext_vector_type(16))) __bf16 v16bf;
typedef __attribute__((ext_vector_type(8)))  float  v8f;
typedef __attribute__((ext_vector_type(4)))  unsigned int u32x4;

union FragU { v16bf v; u32x4 q[2]; };

DEVINL unsigned short f2bf(float x) {
  unsigned int u = __float_as_uint(x);
  u += 0x7FFFu + ((u >> 16) & 1u);     // round-to-nearest-even
  return (unsigned short)(u >> 16);
}
DEVINL float bf2f(unsigned short h) { return __uint_as_float((unsigned int)h << 16); }
DEVINL float sigm(float x) { return 1.f / (1.f + __expf(-x)); }

// A fragment: 16x32 bf16 tile, row-major source, row stride lda (elements).
// lane<16: M=lane, K = {0..7, 16..23}; lane>=16: M=lane-16, K = {8..15, 24..31}
DEVINL v16bf load_a_frag(const unsigned short* __restrict__ base, int lda, int lane) {
  int m  = lane & 15;
  int kb = (lane & 16) >> 1;            // 0 or 8
  const unsigned short* p = base + m * lda + kb;
  FragU f;
  f.q[0] = *(const u32x4*)(p);          // K = kb + 0..7
  f.q[1] = *(const u32x4*)(p + 16);     // K = kb + 16..23
  return f.v;
}
// B fragment: 32(K) x 16(N) where B[k][n] = W[n][k], W row-major stride ldb.
// Streaming weights: single-use per near-cache -> non-temporal hint.
DEVINL v16bf load_b_frag_nt(const unsigned short* __restrict__ base, int ldb, int lane) {
  int n  = lane & 15;
  int kb = (lane & 16) >> 1;            // 0 or 8
  const unsigned short* p = base + n * ldb + kb;
  FragU f;
  f.q[0] = __builtin_nontemporal_load((const u32x4*)p);
  f.q[1] = __builtin_nontemporal_load((const u32x4*)(p + 16));
  return f.v;
}
DEVINL v8f wmma_bf16(v16bf a, v16bf b, v8f c) {
  return __builtin_amdgcn_wmma_f32_16x16x32_bf16(false, a, false, b, (short)0, c, false, false);
}

// Branch-free K-segment GEMM loop: fixed A/W pointers for the whole segment.
DEVINL v8f gemm_seg(v8f acc, const unsigned short* __restrict__ A, int lda,
                    const unsigned short* __restrict__ W, int ldw,
                    int klen, int lane) {
#pragma unroll 2
  for (int kk = 0; kk < klen; kk += 32)
    acc = wmma_bf16(load_a_frag(A + kk, lda, lane),
                    load_b_frag_nt(W + kk, ldw, lane), acc);
  return acc;
}

// ---------------- elementwise / setup kernels ----------------

__global__ void f2bf_kernel(const float* __restrict__ src, unsigned short* __restrict__ dst, int n) {
  for (int i = blockIdx.x * blockDim.x + threadIdx.x; i < n; i += gridDim.x * blockDim.x)
    dst[i] = f2bf(src[i]);
}

// emb_bf[t,b,e] = bf16(embed[tok[b,t], e]);  T=64,B=32,E=512
__global__ void emb_gather_kernel(const float* __restrict__ embed, const int* __restrict__ tok,
                                  unsigned short* __restrict__ emb_bf) {
  int idx = blockIdx.x * blockDim.x + threadIdx.x;   // exact: 64*32*512
  int e  = idx & 511;
  int tb = idx >> 9;
  int b = tb & 31, t = tb >> 5;
  int token = tok[b * 64 + t];
  emb_bf[idx] = f2bf(embed[(size_t)token * 512 + e]);
}

__global__ void state_init_kernel(const float* __restrict__ h0, const float* __restrict__ c0,
                                  unsigned short* __restrict__ h0_bf, unsigned short* __restrict__ h1_bf,
                                  float* __restrict__ c0_st, float* __restrict__ c1_st,
                                  unsigned short* __restrict__ ctx_bf) {
  int idx = blockIdx.x * blockDim.x + threadIdx.x;   // exact: 32*1024
  h0_bf[idx] = f2bf(h0[idx]);
  h1_bf[idx] = f2bf(h0[32768 + idx]);
  c0_st[idx] = c0[idx];
  c1_st[idx] = c0[32768 + idx];
  ctx_bf[idx] = 0;                                   // bf16 zero
}

// ---------------- WMMA gates GEMM: C[32,4096] = [A0|A1|A2] * W^T ----------------
// Each K-segment has a fixed (A, W) pointer pair -> three branch-free loops.
__global__ void gates_kernel(const unsigned short* __restrict__ A0, int lda0, int k0,
                             const unsigned short* __restrict__ W0, int ldw0,
                             const unsigned short* __restrict__ A1, int lda1, int k1,
                             const unsigned short* __restrict__ W1, int ldw1,
                             const unsigned short* __restrict__ A2, int lda2, int k2,
                             const unsigned short* __restrict__ W2, int ldw2,
                             float* __restrict__ Cout) {
  int lane = threadIdx.x & 31;
  int wid  = blockIdx.x * (blockDim.x >> 5) + (threadIdx.x >> 5);  // 512 waves = 256 n-tiles x 2 m-tiles
  int n0 = (wid & 255) * 16;
  int m0 = (wid >> 8) * 16;
  v8f acc = {0.f,0.f,0.f,0.f,0.f,0.f,0.f,0.f};
  acc = gemm_seg(acc, A0 + m0 * lda0, lda0, W0 + n0 * ldw0, ldw0, k0, lane);
  acc = gemm_seg(acc, A1 + m0 * lda1, lda1, W1 + n0 * ldw1, ldw1, k1, lane);
  if (k2)
    acc = gemm_seg(acc, A2 + m0 * lda2, lda2, W2 + n0 * ldw2, ldw2, k2, lane);
  int n  = n0 + (lane & 15);
  int mb = m0 + ((lane & 16) >> 1);
#pragma unroll
  for (int r = 0; r < 8; ++r)
    Cout[(mb + r) * 4096 + n] = acc[r];
}

// ---------------- LSTM pointwise update (bias folded here) ----------------
__global__ void lstm_update_kernel(const float* __restrict__ gates, const float* __restrict__ bias,
                                   float* __restrict__ c_state, unsigned short* __restrict__ h_bf,
                                   unsigned short* __restrict__ dec_out) {
  int idx = blockIdx.x * blockDim.x + threadIdx.x;   // exact: 32*1024
  int b = idx >> 10, j = idx & 1023;
  const float* g = gates + b * 4096;
  float gi = g[j]        + bias[j];
  float gf = g[1024 + j] + bias[1024 + j];
  float gc = g[2048 + j] + bias[2048 + j];
  float go = g[3072 + j] + bias[3072 + j];
  float c = sigm(gf) * c_state[idx] + sigm(gi) * tanhf(gc);
  c_state[idx] = c;
  unsigned short hb = f2bf(sigm(go) * tanhf(c));
  h_bf[idx] = hb;
  if (dec_out) dec_out[idx] = hb;
}

// ---------------- attention: scores -> masked softmax -> context ----------------
__global__ void attention_kernel(const float* __restrict__ encoded, const int* __restrict__ enc_lens,
                                 const unsigned short* __restrict__ h1_bf,
                                 unsigned short* __restrict__ ctx_bf,
                                 unsigned short* __restrict__ ctxs_out) {
  __shared__ float part[256];
  __shared__ float sc[64];
  __shared__ float snorm[2];
  int b = blockIdx.x, tid = threadIdx.x;
  int s = tid >> 2, q = tid & 3;
  const float* e = encoded + ((size_t)b * 64 + s) * 1024 + q * 256;
  const unsigned short* h = h1_bf + b * 1024 + q * 256;
  float acc = 0.f;
#pragma unroll 4
  for (int k = 0; k < 256; ++k) acc += e[k] * bf2f(h[k]);
  part[tid] = acc;
  __syncthreads();
  if (tid < 64) {
    float v = part[tid*4] + part[tid*4+1] + part[tid*4+2] + part[tid*4+3];
    sc[tid] = (tid < enc_lens[b]) ? v : -__builtin_inff();
  }
  __syncthreads();
  if (tid == 0) {
    float m = -__builtin_inff();
    for (int i = 0; i < 64; ++i) m = fmaxf(m, sc[i]);
    float sum = 0.f;
    for (int i = 0; i < 64; ++i) sum += __expf(sc[i] - m);
    snorm[0] = m; snorm[1] = sum;
  }
  __syncthreads();
  if (tid < 64) sc[tid] = __expf(sc[tid] - snorm[0]) / snorm[1];
  __syncthreads();
  for (int j = tid; j < 1024; j += 256) {
    float a = 0.f;
    for (int s2 = 0; s2 < 64; ++s2)
      a += sc[s2] * encoded[((size_t)b * 64 + s2) * 1024 + j];
    unsigned short v = f2bf(a);
    ctx_bf[b * 1024 + j]   = v;
    ctxs_out[b * 1024 + j] = v;
  }
}

// ---------------- output projection: hh = tanh([decs|ctxs] @ Wout^T + b) -> bf16 ----------------
__global__ void outproj_kernel(const unsigned short* __restrict__ decs,
                               const unsigned short* __restrict__ ctxs,
                               const unsigned short* __restrict__ Wout_bf,
                               const float* __restrict__ b_out,
                               unsigned short* __restrict__ hh_bf) {
  int lane = threadIdx.x & 31;
  int wid  = blockIdx.x * (blockDim.x >> 5) + (threadIdx.x >> 5);  // 4096 waves = 32 n x 128 m
  int n0 = (wid & 31) * 16;
  int m0 = (wid >> 5) * 16;
  v8f acc = {0.f,0.f,0.f,0.f,0.f,0.f,0.f,0.f};
  acc = gemm_seg(acc, decs + m0 * 1024, 1024, Wout_bf + n0 * 2048,        2048, 1024, lane);
  acc = gemm_seg(acc, ctxs + m0 * 1024, 1024, Wout_bf + n0 * 2048 + 1024, 2048, 1024, lane);
  int n  = n0 + (lane & 15);
  int mb = m0 + ((lane & 16) >> 1);
  float bias = b_out[n];
#pragma unroll
  for (int r = 0; r < 8; ++r)
    hh_bf[(mb + r) * 512 + n] = f2bf(tanhf(acc[r] + bias));
}

// ---------------- fused logits GEMM + streaming logsumexp + target gather ----------------
// One block per 16-row tile; 8 waves stream disjoint 16-column vocab slices.
__global__ void logits_lse_kernel(const unsigned short* __restrict__ hh_bf,
                                  const unsigned short* __restrict__ embed_bf,
                                  const int* __restrict__ tok,
                                  float* __restrict__ rowLse,
                                  float* __restrict__ rowTgt) {
  __shared__ float red[8][16][2];
  int lane = threadIdx.x & 31;
  int wave = threadIdx.x >> 5;
  int m0   = blockIdx.x * 16;
  int hi8  = (lane & 16) >> 1;
  v16bf afrag[16];
#pragma unroll
  for (int kt = 0; kt < 16; ++kt)
    afrag[kt] = load_a_frag(hh_bf + m0 * 512 + kt * 32, 512, lane);
  int labels[8];
#pragma unroll
  for (int r = 0; r < 8; ++r) {
    int m = m0 + hi8 + r;
    int t = m >> 5, b = m & 31;                 // m = t*B + b
    labels[r] = (t + 1 < 64) ? tok[b * 64 + t + 1] : 0;
  }
  float mx[8], sm[8];
#pragma unroll
  for (int r = 0; r < 8; ++r) { mx[r] = -__builtin_inff(); sm[r] = 0.f; }

  for (int v0 = wave * 16; v0 < 32000; v0 += 128) {
    v8f acc = {0.f,0.f,0.f,0.f,0.f,0.f,0.f,0.f};
#pragma unroll
    for (int kt = 0; kt < 16; ++kt)
      acc = wmma_bf16(afrag[kt],
                      load_b_frag_nt(embed_bf + (size_t)v0 * 512 + kt * 32, 512, lane), acc);
    int col = v0 + (lane & 15);
#pragma unroll
    for (int r = 0; r < 8; ++r) {
      float val = acc[r];
      if (col == labels[r]) rowTgt[m0 + hi8 + r] = val;   // exactly one writer per row
      if (val <= mx[r]) {
        sm[r] += __expf(val - mx[r]);
      } else {
        sm[r] = sm[r] * __expf(mx[r] - val) + 1.f;
        mx[r] = val;
      }
    }
  }
  // merge across the 16 lanes of each half-wave (same rows within a half)
#pragma unroll
  for (int r = 0; r < 8; ++r) {
#pragma unroll
    for (int off = 8; off >= 1; off >>= 1) {
      float om = __shfl_xor(mx[r], off, 32);
      float os = __shfl_xor(sm[r], off, 32);
      float nm = fmaxf(mx[r], om);
      sm[r] = sm[r] * __expf(mx[r] - nm) + os * __expf(om - nm);
      mx[r] = nm;
    }
  }
  if ((lane & 15) == 0) {
#pragma unroll
    for (int r = 0; r < 8; ++r) { red[wave][hi8 + r][0] = mx[r]; red[wave][hi8 + r][1] = sm[r]; }
  }
  __syncthreads();
  if (threadIdx.x < 16) {
    int row = threadIdx.x;
    float M = -__builtin_inff(), S = 0.f;
#pragma unroll
    for (int w = 0; w < 8; ++w) {
      float m_ = red[w][row][0], s_ = red[w][row][1];
      float nm = fmaxf(M, m_);
      S = S * __expf(M - nm) + s_ * __expf(m_ - nm);
      M = nm;
    }
    rowLse[m0 + row] = M + __logf(S);
  }
}

// ---------------- final masked loss reduction ----------------
__global__ void loss_kernel(const float* __restrict__ lse, const float* __restrict__ tgt,
                            const int* __restrict__ tlen, float* __restrict__ out) {
  __shared__ float red[256];
  float acc = 0.f;
  for (int m = (int)threadIdx.x; m < 2048; m += 256) {
    int t = m >> 5, b = m & 31;
    if (t < tlen[b] - 1) acc += lse[m] - tgt[m];
  }
  red[threadIdx.x] = acc;
  __syncthreads();
  for (int s = 128; s > 0; s >>= 1) {
    if ((int)threadIdx.x < s) red[threadIdx.x] += red[threadIdx.x + s];
    __syncthreads();
  }
  if (threadIdx.x == 0) out[0] = red[0];
}

// ---------------- host launcher ----------------
extern "C" void kernel_launch(void* const* d_in, const int* in_sizes, int n_in,
                              void* d_out, int out_size, void* d_ws, size_t ws_size,
                              hipStream_t stream) {
  (void)in_sizes; (void)n_in; (void)out_size; (void)ws_size;
  const int*   tok   = (const int*)d_in[0];
  const int*   tlen  = (const int*)d_in[1];
  const float* enc   = (const float*)d_in[2];
  const int*   elen  = (const int*)d_in[3];
  const float* h0    = (const float*)d_in[4];
  const float* c0    = (const float*)d_in[5];
  const float* emb   = (const float*)d_in[6];
  const float* Wih0  = (const float*)d_in[7];
  const float* Whh0  = (const float*)d_in[8];
  const float* b0    = (const float*)d_in[9];
  const float* Wih1  = (const float*)d_in[10];
  const float* Whh1  = (const float*)d_in[11];
  const float* b1    = (const float*)d_in[12];
  const float* Wout  = (const float*)d_in[13];
  const float* bout  = (const float*)d_in[14];
  float* out = (float*)d_out;

  char* wsp = (char*)d_ws;
  auto carve = [&](size_t bytes) -> void* {
    void* p = (void*)wsp;
    wsp += (bytes + 255) & ~(size_t)255;
    return p;
  };
  unsigned short* emb_bf   = (unsigned short*)carve(2ull * 64 * 32 * 512);
  unsigned short* Wih0_bf  = (unsigned short*)carve(2ull * 4096 * 1536);
  unsigned short* Whh0_bf  = (unsigned short*)carve(2ull * 4096 * 1024);
  unsigned short* Wih1_bf  = (unsigned short*)carve(2ull * 4096 * 1024);
  unsigned short* Whh1_bf  = (unsigned short*)carve(2ull * 4096 * 1024);
  unsigned short* Wout_bf  = (unsigned short*)carve(2ull * 512 * 2048);
  unsigned short* embed_bf = (unsigned short*)carve(2ull * 32000 * 512);
  unsigned short* h0_bf    = (unsigned short*)carve(2ull * 32 * 1024);
  unsigned short* h1_bf    = (unsigned short*)carve(2ull * 32 * 1024);
  unsigned short* ctx_bf   = (unsigned short*)carve(2ull * 32 * 1024);
  float*          c0_st    = (float*)carve(4ull * 32 * 1024);
  float*          c1_st    = (float*)carve(4ull * 32 * 1024);
  float*          gates    = (float*)carve(4ull * 32 * 4096);
  unsigned short* decs_bf  = (unsigned short*)carve(2ull * 64 * 32 * 1024);
  unsigned short* ctxs_bf  = (unsigned short*)carve(2ull * 64 * 32 * 1024);
  unsigned short* hh_bf    = (unsigned short*)carve(2ull * 2048 * 512);
  float*          rowLse   = (float*)carve(4ull * 2048);
  float*          rowTgt   = (float*)carve(4ull * 2048);

  // --- precision conversion / gathers (embed fits L2 in bf16: 32.8 MB) ---
  f2bf_kernel<<<2048, 256, 0, stream>>>(Wih0, Wih0_bf, 4096 * 1536);
  f2bf_kernel<<<2048, 256, 0, stream>>>(Whh0, Whh0_bf, 4096 * 1024);
  f2bf_kernel<<<2048, 256, 0, stream>>>(Wih1, Wih1_bf, 4096 * 1024);
  f2bf_kernel<<<2048, 256, 0, stream>>>(Whh1, Whh1_bf, 4096 * 1024);
  f2bf_kernel<<<1024, 256, 0, stream>>>(Wout, Wout_bf, 512 * 2048);
  f2bf_kernel<<<4096, 256, 0, stream>>>(emb, embed_bf, 32000 * 512);
  emb_gather_kernel<<<4096, 256, 0, stream>>>(emb, tok, emb_bf);
  state_init_kernel<<<128, 256, 0, stream>>>(h0, c0, h0_bf, h1_bf, c0_st, c1_st, ctx_bf);

  // --- sequential decode: 64 dependent steps, weights stay resident in L2 ---
  for (int t = 0; t < 64; ++t) {
    // cell0: A = [x_t (512) | ctx (1024) | h0 (1024)], W = [W_ih0[:, :512] | W_ih0[:, 512:] | W_hh0]
    gates_kernel<<<128, 128, 0, stream>>>(
        emb_bf + (size_t)t * 32 * 512, 512, 512,  Wih0_bf,       1536,
        ctx_bf,                        1024, 1024, Wih0_bf + 512, 1536,
        h0_bf,                         1024, 1024, Whh0_bf,       1024,
        gates);
    lstm_update_kernel<<<128, 256, 0, stream>>>(gates, b0, c0_st, h0_bf, (unsigned short*)nullptr);
    // cell1: A = [h0_new (1024) | h1 (1024)], W = [W_ih1 | W_hh1]
    gates_kernel<<<128, 128, 0, stream>>>(
        h0_bf, 1024, 1024, Wih1_bf, 1024,
        h1_bf, 1024, 1024, Whh1_bf, 1024,
        h1_bf, 1024, 0,    Whh1_bf, 1024,
        gates);
    lstm_update_kernel<<<128, 256, 0, stream>>>(gates, b1, c1_st, h1_bf,
                                                decs_bf + (size_t)t * 32 * 1024);
    attention_kernel<<<32, 256, 0, stream>>>(enc, elen, h1_bf, ctx_bf,
                                             ctxs_bf + (size_t)t * 32 * 1024);
  }

  // --- parallel tail ---
  outproj_kernel<<<512, 256, 0, stream>>>(decs_bf, ctxs_bf, Wout_bf, bout, hh_bf);
  logits_lse_kernel<<<128, 256, 0, stream>>>(hh_bf, embed_bf, tok, rowLse, rowTgt);
  loss_kernel<<<1, 256, 0, stream>>>(rowLse, rowTgt, tlen, out);
}
